// discrete_net_model_52725018526328
// MI455X (gfx1250) — matrix operand order repaired and checked by reading, not compiled, not run on hardware
//
#include <hip/hip_runtime.h>

// ---------------------------------------------------------------------------
// 2D-LSTM scan for MI455X (gfx1250, wave32, WMMA).
// One persistent workgroup per batch (B=32). 127 sequential steps in-kernel.
// bf16 WMMA (16x16x32, f32 accum); weights pre-packed to fragment order in ws.
// x@Wx is folded into the WMMA as a 5th K-chunk (hbuf cols 128..130 carry x_t).
// ---------------------------------------------------------------------------

typedef __bf16 bf16_t;
typedef __attribute__((ext_vector_type(16))) __bf16        v16bf;
typedef __attribute__((ext_vector_type(8)))  float         v8f;
typedef __attribute__((ext_vector_type(4)))  unsigned int  v4u;

union FragU { v4u q[2]; v16bf v; };

__device__ __forceinline__ v8f wmma_bf16(v16bf a, v16bf b, v8f c) {
#if defined(__HIP_DEVICE_COMPILE__)
  return __builtin_amdgcn_wmma_f32_16x16x32_bf16(false, a, false, b,
                                                 (short)0, c, false, false);
#else
  (void)a; (void)b;
  return c;
#endif
}

// A-fragment (16x32 bf16, M x K): lane holds row M=lane&15; elements 0..7 are
// K = kb..kb+7, elements 8..15 are K = kb+16..kb+23, with kb = 32*kk + 8*(lane>>4).
__device__ __forceinline__ v16bf load_fragA(const bf16_t* rowp, int kb) {
  FragU f;
  f.q[0] = *(const v4u*)(rowp + kb);
  f.q[1] = *(const v4u*)(rowp + kb + 16);
  return f.v;
}

// B-fragment pre-packed contiguously: 32 bytes per lane.
__device__ __forceinline__ v16bf load_fragB(const bf16_t* p) {
  FragU f;
  f.q[0] = *(const v4u*)(p);
  f.q[1] = *(const v4u*)(p + 8);
  return f.v;
}

__device__ __forceinline__ float sigm(float x) { return 1.0f / (1.0f + __expf(-x)); }

#define NB       32
#define TT       128
#define TSTEPS   127
#define SS       64
#define HH       128
#define NG       512     // 4*H
#define OO       16
#define TPRED    4
#define TOUT     123
#define HSTR     168     // hbuf row stride: 128 h + 32 x-chunk + 8 pad (bank spread)
#define MATSZ    65536   // packed elements per [128x512] matrix
#define WXOFF    (3 * MATSZ)            // 32 tiles: Wx as K-chunk 128..159 (zero padded)
#define WOOFF    (WXOFF + 32 * 512)     // 4 tiles: Wout
#define PACKN    (WOOFF + 4 * 512)

// ---------------------------------------------------------------------------
// Pack Wh/Wa/Wb (f32 [128,512]), Wx (f32 [3,512]) and Wout (f32 [128,16]) into
// bf16 B-fragment order. B-layout inside a 32xK-chunk x 16 tile:
//   element addr = (tile*32 + lane)*16 + e,  Kloc = 8*(lane>>4) + (e&7) + 16*(e>>3),
//   col = 16*nt + (lane&15).
// ---------------------------------------------------------------------------
__global__ void pack_weights_kernel(const float* __restrict__ Wh,
                                    const float* __restrict__ Wa,
                                    const float* __restrict__ Wb,
                                    const float* __restrict__ Wx,
                                    const float* __restrict__ Wout,
                                    unsigned short* __restrict__ ws) {
  const int d = blockIdx.x * 256 + threadIdx.x;
  float v;
  if (d < 3 * MATSZ) {
    const int m    = d >> 16;
    const int r    = d & 65535;
    const int tile = r >> 9;          // nt*4 + kk
    const int lane = (r >> 4) & 31;
    const int e    = r & 15;
    const int nt   = tile >> 2;
    const int kk   = tile & 3;
    const int K    = 32 * kk + 8 * (lane >> 4) + (e & 7) + 16 * (e >> 3);
    const int col  = 16 * nt + (lane & 15);
    const float* W = (m == 0) ? Wh : ((m == 1) ? Wa : Wb);
    v = W[K * NG + col];
  } else if (d < WOOFF) {
    const int r    = d - WXOFF;
    const int nt   = r >> 9;          // 0..31
    const int lane = (r >> 4) & 31;
    const int e    = r & 15;
    const int Kl   = 8 * (lane >> 4) + (e & 7) + 16 * (e >> 3);   // 0..31 within x-chunk
    const int col  = 16 * nt + (lane & 15);
    v = (Kl < 3) ? Wx[Kl * NG + col] : 0.0f;
  } else if (d < PACKN) {
    const int r    = d - WOOFF;
    const int kk   = r >> 9;          // 0..3
    const int lane = (r >> 4) & 31;
    const int e    = r & 15;
    const int K    = 32 * kk + 8 * (lane >> 4) + (e & 7) + 16 * (e >> 3);
    const int col  = lane & 15;
    v = Wout[K * OO + col];
  } else {
    return;
  }
  const unsigned u   = __float_as_uint(v);
  const unsigned rnd = (u + 0x7FFFu + ((u >> 16) & 1u)) >> 16;  // RNE to bf16
  ws[d] = (unsigned short)rnd;
}

// ---------------------------------------------------------------------------
// Main recurrent kernel: 32 blocks x 256 threads (8 wave32).
// Wave w = (mt = w>>1) x (g = w&1): owns s-rows [16mt,16mt+16) and
// h-cols [64g, 64g+64) => gate tiles nt_global = gate*8 + (4g..4g+3).
// ---------------------------------------------------------------------------
__global__ __launch_bounds__(256)
void lstm2d_scan_kernel(const float* __restrict__ xall,   // [B,T,S,3]
                        const float* __restrict__ bvec,   // [512]
                        const float* __restrict__ bout,   // [16]
                        const bf16_t* __restrict__ wpk,   // packed weights
                        float* __restrict__ out)          // [B,123,64,16]
{
  __shared__ alignas(16) bf16_t hbuf[2][66 * HSTR];  // rows 0/65 and cols 131..159 stay 0
  __shared__ float bl[NG];
  __shared__ float boutl[OO];

  const int b     = blockIdx.x;
  const int tid   = threadIdx.x;
  const int lane  = tid & 31;
  const int wv    = tid >> 5;
  const int laneM = lane & 15;
  const int half  = lane >> 4;
  const int mt    = wv >> 1;
  const int g     = wv & 1;

  for (int i = tid; i < 66 * HSTR; i += 256) {
    hbuf[0][i] = (bf16_t)0.0f;
    hbuf[1][i] = (bf16_t)0.0f;
  }
  for (int i = tid; i < NG; i += 256) bl[i] = bvec[i];
  if (tid < OO) boutl[tid] = bout[tid];

  // Cell state in registers: creg[ntp][r] <-> (s = 16mt+r+8half, j = 16*(4g+ntp)+laneM)
  float creg[4][8];
#pragma unroll
  for (int n = 0; n < 4; ++n)
#pragma unroll
    for (int r = 0; r < 8; ++r) creg[n][r] = 0.0f;

  const int arow = 16 * mt + laneM + 1;  // center h row in hbuf for this lane's A rows

  int cur = 0;
  for (int t = 0; t < TSTEPS; ++t) {
    // Stage x_t directly into the read buffer's x-chunk (cols 128..130, bf16-exact)
    if (tid < SS * 3) {
      const int s_ = tid / 3;
      const int i_ = tid - 3 * s_;
      hbuf[cur][(size_t)(s_ + 1) * HSTR + 128 + i_] =
          (bf16_t)xall[(((size_t)b * TT + t) * SS) * 3 + tid];
    }
    __syncthreads();

    const bf16_t* hc = &hbuf[cur][0];
    bf16_t*       hn = &hbuf[cur ^ 1][0];

#pragma unroll
    for (int ntp = 0; ntp < 4; ++ntp) {
      const int ntl = 4 * g + ntp;  // h column tile 0..7

      // C init = bias splat (x@Wx comes via the 5th K-chunk WMMA)
      v8f acc[4];
#pragma unroll
      for (int gate = 0; gate < 4; ++gate) {
        const float bb = bl[gate * 128 + ntl * 16 + laneM];
#pragma unroll
        for (int r = 0; r < 8; ++r) acc[gate][r] = bb;
      }

      // Wave-uniform tile base (SGPR); per-lane part is a fixed +lane*16.
      // Fragment offsets per (gate, matrix) are compile-time immediates.
      const bf16_t* pkk = wpk + ((size_t)ntl * 4) * 512;  // advanced by 512/kk

#pragma unroll 1  // keep dynamic: bounds live fragments, avoids spills
      for (int kk = 0; kk < 4; ++kk) {
        const int kb = 32 * kk + 8 * half;
        const v16bf Ac = load_fragA(hc + (size_t)(arow    ) * HSTR, kb);  // h
        const v16bf Aa = load_fragA(hc + (size_t)(arow + 1) * HSTR, kb);  // h_after
        const v16bf Ab = load_fragA(hc + (size_t)(arow - 1) * HSTR, kb);  // h_before

        // Issue all 12 B-fragment loads (24 x global_load_b128) up front;
        // they return in order so WMMAs below overlap the tail of the loads.
        const bf16_t* lp = pkk + (size_t)lane * 16;
        v16bf Bf[12];
#pragma unroll
        for (int gate = 0; gate < 4; ++gate) {
          // gate stride: 8 nt-tiles * 4 kk * 512 = 16384 elements
          Bf[gate * 3 + 0] = load_fragB(lp + gate * 16384);
          Bf[gate * 3 + 1] = load_fragB(lp + gate * 16384 + MATSZ);
          Bf[gate * 3 + 2] = load_fragB(lp + gate * 16384 + 2 * MATSZ);
        }
#pragma unroll
        for (int gate = 0; gate < 4; ++gate) {
          acc[gate] = wmma_bf16(Ac, Bf[gate * 3 + 0], acc[gate]);  // Wh
          acc[gate] = wmma_bf16(Aa, Bf[gate * 3 + 1], acc[gate]);  // Wa
          acc[gate] = wmma_bf16(Ab, Bf[gate * 3 + 2], acc[gate]);  // Wb
        }
        pkk += 512;
      }
      {
        // x contribution: K-chunk 128..159 of the center rows x packed Wx tiles
        const v16bf Ax = load_fragA(hc + (size_t)arow * HSTR, 128 + 8 * half);
        const bf16_t* px = wpk + WXOFF + (size_t)ntl * 512 + (size_t)lane * 16;
        v16bf Bx[4];
#pragma unroll
        for (int gate = 0; gate < 4; ++gate)
          Bx[gate] = load_fragB(px + gate * 4096);  // 8 nt-tiles * 512
#pragma unroll
        for (int gate = 0; gate < 4; ++gate)
          acc[gate] = wmma_bf16(Ax, Bx[gate], acc[gate]);
      }

      // Elementwise LSTM update in registers; write h_new (bf16) to ping buffer
#pragma unroll
      for (int r = 0; r < 8; ++r) {
        const float ig = sigm(acc[0][r]);
        const float fg = sigm(acc[1][r]);
        const float gg = tanhf(acc[2][r]);
        const float og = sigm(acc[3][r]);
        const float cn = fg * creg[ntp][r] + ig * gg;
        creg[ntp][r] = cn;
        const float hv = og * tanhf(cn);
        const int srow = 16 * mt + r + 8 * half;
        hn[(size_t)(srow + 1) * HSTR + ntl * 16 + laneM] = (bf16_t)hv;
      }
    }
    __syncthreads();

    // Output projection: out = h_new @ Wout + bout (waves with g==0, one mt each)
    if (t >= TPRED && g == 0) {
      v8f oacc;
      const float bo = boutl[laneM];
#pragma unroll
      for (int r = 0; r < 8; ++r) oacc[r] = bo;
      const bf16_t* po = wpk + WOOFF + (size_t)lane * 16;
      v16bf Bo[4];
#pragma unroll
      for (int kk = 0; kk < 4; ++kk) Bo[kk] = load_fragB(po + kk * 512);
#pragma unroll
      for (int kk = 0; kk < 4; ++kk) {
        const v16bf A = load_fragA(hn + (size_t)arow * HSTR, 32 * kk + 8 * half);
        oacc = wmma_bf16(A, Bo[kk], oacc);
      }
      float* ob = out + (((size_t)b * TOUT + (t - TPRED)) * SS) * OO;
#pragma unroll
      for (int r = 0; r < 8; ++r) {
        const int srow = 16 * mt + r + 8 * half;
        ob[srow * OO + laneM] = oacc[r];
      }
    }
    cur ^= 1;
  }
}

// ---------------------------------------------------------------------------
// delta_N: per (b,t',s): 31 windowed dot products of outputs row with
// input_proba row (one-hot at s==0, previous-s outputs otherwise). HBM-trivial.
// ---------------------------------------------------------------------------
__global__ void delta_kernel(const float* __restrict__ xall,
                             const float* __restrict__ outputs,  // [B,123,64,16]
                             float* __restrict__ delta)          // [B,123,64,31]
{
  const int id = blockIdx.x * 256 + threadIdx.x;
  const int ROWS = NB * TOUT * SS;
  if (id >= ROWS) return;
  const int s  = id & (SS - 1);
  const int bt = id >> 6;
  const int tp = bt % TOUT;
  const int b  = bt / TOUT;

  float ov[OO], p[OO];
  const float* o = outputs + (size_t)id * OO;
#pragma unroll
  for (int k = 0; k < OO; ++k) ov[k] = o[k];

  if (s == 0) {
    const int cls = (int)xall[(((size_t)b * TT + (tp + TPRED)) * SS) * 3];
#pragma unroll
    for (int k = 0; k < OO; ++k) p[k] = (k == cls) ? 1.0f : 0.0f;
  } else {
    const float* pm = outputs + ((size_t)id - 1) * OO;
#pragma unroll
    for (int k = 0; k < OO; ++k) p[k] = pm[k];
  }

  float* dl = delta + (size_t)id * (2 * OO - 1);
#pragma unroll
  for (int l = 0; l < 2 * OO - 1; ++l) {
    const int k0 = (l < OO - 1) ? (OO - 1 - l) : 0;
    const int k1 = (l > OO - 1) ? (2 * OO - 2 - l) : (OO - 1);
    float sum = 0.0f;
    for (int k = k0; k <= k1; ++k) sum += ov[l + k - (OO - 1)] * p[k];
    dl[l] = sum;
  }
}

// ---------------------------------------------------------------------------
extern "C" void kernel_launch(void* const* d_in, const int* in_sizes, int n_in,
                              void* d_out, int out_size, void* d_ws, size_t ws_size,
                              hipStream_t stream) {
  (void)in_sizes; (void)n_in; (void)out_size; (void)ws_size;
  const float* input = (const float*)d_in[0];  // [32,128,64,3]
  const float* Wx    = (const float*)d_in[1];  // [3,512]
  const float* Wh    = (const float*)d_in[2];  // [128,512]
  const float* Wa    = (const float*)d_in[3];  // [128,512]
  const float* Wb    = (const float*)d_in[4];  // [128,512]
  const float* bv    = (const float*)d_in[5];  // [512]
  const float* Wout  = (const float*)d_in[6];  // [128,16]
  const float* bout  = (const float*)d_in[7];  // [16]

  float* outputs = (float*)d_out;                          // [32,123,64,16]
  float* delta   = outputs + (size_t)NB * TOUT * SS * OO;  // [32,123,64,31]
  unsigned short* ws = (unsigned short*)d_ws;              // PACKN bf16

  pack_weights_kernel<<<(PACKN + 255) / 256, 256, 0, stream>>>(Wh, Wa, Wb, Wx, Wout, ws);
  lstm2d_scan_kernel<<<NB, 256, 0, stream>>>(input, bv, bout, (const bf16_t*)ws, outputs);
  const int rows = NB * TOUT * SS;
  delta_kernel<<<(rows + 255) / 256, 256, 0, stream>>>(input, outputs, delta);
}
